// MixtureOfExperts_23201413333467
// MI455X (gfx1250) — matrix operand order repaired and checked by reading, not compiled, not run on hardware
//
#include <hip/hip_runtime.h>

// MoE: N=8192 tokens, D=2048, E=8 experts, top-2 gating.
// gate -> per-expert token lists -> gathered bf16 WMMA GEMM per expert
// (top-2 work only: ~137 GFLOP vs 550 dense), fp32 accumulate,
// scatter-add epilogue (exactly 2 contributions/element -> deterministic).
// GEMM: 128x256 block tile, 8 waves as 2x4, 64x64 per wave, BK=64,
// double-buffered LDS fed by GLOBAL_LOAD_ASYNC_TO_LDS (ASYNCcnt) when
// available, register staging otherwise.

#define NTOK 8192
#define DDIM 2048
#define NEXP 8
#define BM 128
#define BN 256
#define BK 64
#define KT2 (DDIM / BK)

#if defined(__has_builtin)
#if __has_builtin(__builtin_amdgcn_global_load_async_to_lds_b128)
#define HAVE_ASYNC_LDS 1
#endif
#endif

typedef __attribute__((ext_vector_type(16))) __bf16 v16bf;
typedef __attribute__((ext_vector_type(8)))  __bf16 v8bf;
typedef __attribute__((ext_vector_type(8)))  float  v8f;
typedef int v4i __attribute__((vector_size(16)));   // matches builtin param type

static __device__ __forceinline__ unsigned short f2bf(float f) {
  union { float f; unsigned int u; } c; c.f = f;
  unsigned int u = c.u;
  u += 0x7FFFu + ((u >> 16) & 1u);   // round-to-nearest-even
  return (unsigned short)(u >> 16);
}

// ---------------- fp32 -> bf16 conversion (vectorized) ----------------------
__global__ void cvt_f32_bf16(const float* __restrict__ in,
                             unsigned short* __restrict__ out, int n4) {
  int i = blockIdx.x * blockDim.x + threadIdx.x;
  if (i >= n4) return;
  float4 v = reinterpret_cast<const float4*>(in)[i];
  ushort4 o;
  o.x = f2bf(v.x); o.y = f2bf(v.y); o.z = f2bf(v.z); o.w = f2bf(v.w);
  reinterpret_cast<ushort4*>(out)[i] = o;
}

// ---------------- gating: one wave32 per token ------------------------------
__global__ void moe_gate(const float* __restrict__ x, const float* __restrict__ Wg,
                         const float* __restrict__ bg, int* __restrict__ counts,
                         int* __restrict__ tlist, float* __restrict__ wlist) {
  const int token = blockIdx.x * 8 + (threadIdx.x >> 5);
  const int lane  = threadIdx.x & 31;
  float acc[NEXP];
#pragma unroll
  for (int e = 0; e < NEXP; ++e) acc[e] = 0.f;
  const float* xr = x + (size_t)token * DDIM;
  for (int k = lane; k < DDIM; k += 32) {
    float xv = xr[k];
#pragma unroll
    for (int e = 0; e < NEXP; ++e) acc[e] += xv * Wg[e * DDIM + k];
  }
#pragma unroll
  for (int off = 16; off >= 1; off >>= 1) {
#pragma unroll
    for (int e = 0; e < NEXP; ++e) acc[e] += __shfl_xor(acc[e], off, 32);
  }
  if (lane == 0) {
    float lg[NEXP], p[NEXP];
    float m = -3.4e38f;
#pragma unroll
    for (int e = 0; e < NEXP; ++e) { lg[e] = acc[e] + bg[e]; m = fmaxf(m, lg[e]); }
    float s = 0.f;
#pragma unroll
    for (int e = 0; e < NEXP; ++e) { p[e] = expf(lg[e] - m); s += p[e]; }
    const float inv = 1.f / s;
    int i0 = 0;
#pragma unroll
    for (int e = 1; e < NEXP; ++e) if (p[e] > p[i0]) i0 = e;
    int i1 = (i0 == 0) ? 1 : 0;
#pragma unroll
    for (int e = 0; e < NEXP; ++e) if (e != i0 && p[e] > p[i1]) i1 = e;
    int s0 = atomicAdd(&counts[i0], 1);
    tlist[i0 * NTOK + s0] = token; wlist[i0 * NTOK + s0] = p[i0] * inv;
    int s1 = atomicAdd(&counts[i1], 1);
    tlist[i1 * NTOK + s1] = token; wlist[i1 * NTOK + s1] = p[i1] * inv;
  }
}

// ---------------- per-expert gathered GEMM (bf16 WMMA, fp32 acc) ------------
__global__ __launch_bounds__(256) void moe_expert_gemm(
    const unsigned short* __restrict__ xbf,   // [NTOK, D] bf16
    const unsigned short* __restrict__ wbf,   // [E, D, D] bf16
    const float* __restrict__ bexp,           // [E, D]
    const int* __restrict__ counts,
    const int* __restrict__ tlist,            // [E, NTOK]
    const float* __restrict__ wlist,          // [E, NTOK]
    float* __restrict__ out)                  // [NTOK, D] fp32 (pre-zeroed)
{
  const int e   = blockIdx.z;
  const int cnt = counts[e];
  const int m0  = blockIdx.y * BM;
  if (m0 >= cnt) return;
  const int n0  = blockIdx.x * BN;

  __shared__ alignas(32) unsigned short As[2][BM * BK];   // 2 x 16 KB
  __shared__ alignas(32) unsigned short Bs[2][BN * BK];   // 2 x 32 KB
  __shared__ int   toks[BM];
  __shared__ float wts[BM];

  const int tid = threadIdx.x;
  if (tid < BM) {
    int idx = m0 + tid;
    if (idx < cnt) { toks[tid] = tlist[e * NTOK + idx]; wts[tid] = wlist[e * NTOK + idx]; }
    else           { toks[tid] = 0;                     wts[tid] = 0.f; }
  }
  __syncthreads();

  // ---- hoisted gather pointers: per thread 4 A chunks + 8 B chunks (16B) --
  const int arow = tid >> 3;           // 0..31; row groups step by 32
  const int jj   = (tid & 7) * 8;      // bf16 offset of 16B chunk within BK
  const size_t wbase = (size_t)e * DDIM * DDIM;
  const unsigned short* aP[4];
  const unsigned short* bP[8];
#pragma unroll
  for (int g = 0; g < 4; ++g)
    aP[g] = xbf + (size_t)toks[arow + 32 * g] * DDIM + jj;
#pragma unroll
  for (int g = 0; g < 8; ++g)
    bP[g] = wbf + wbase + (size_t)(n0 + arow + 32 * g) * DDIM + jj;
  const int aOff = arow * BK + jj;     // LDS element offsets; +32*g rows
  const int bOff = arow * BK + jj;

  const int lane = tid & 31;
  const int wave = tid >> 5;
  const int wm   = wave & 1;           // 2 wave-rows of 64
  const int wn   = wave >> 1;          // 4 wave-cols of 64
  const int l16  = lane & 15;
  const bool hi  = lane >= 16;
  const int akc  = hi ? 8 : 0;
  const int bkc  = hi ? 8 : 0;

  v8f acc[4][4];
#pragma unroll
  for (int mi = 0; mi < 4; ++mi)
#pragma unroll
    for (int ni = 0; ni < 4; ++ni) acc[mi][ni] = {};

  // stage one BK-slice (A:128x64, B:256x64) of tile `buf` at k-offset ko
  auto stage = [&](int buf, int ko) {
#ifdef HAVE_ASYNC_LDS
#pragma unroll
    for (int g = 0; g < 4; ++g) {
      __builtin_amdgcn_global_load_async_to_lds_b128(
          (__attribute__((address_space(1))) v4i*)(aP[g] + ko),
          (__attribute__((address_space(3))) v4i*)&As[buf][aOff + g * 32 * BK],
          0, 0);
    }
#pragma unroll
    for (int g = 0; g < 8; ++g) {
      __builtin_amdgcn_global_load_async_to_lds_b128(
          (__attribute__((address_space(1))) v4i*)(bP[g] + ko),
          (__attribute__((address_space(3))) v4i*)&Bs[buf][bOff + g * 32 * BK],
          0, 0);
    }
#else
#pragma unroll
    for (int g = 0; g < 4; ++g) {
      int4 t = *reinterpret_cast<const int4*>(aP[g] + ko);
      *reinterpret_cast<int4*>(&As[buf][aOff + g * 32 * BK]) = t;
    }
#pragma unroll
    for (int g = 0; g < 8; ++g) {
      int4 t = *reinterpret_cast<const int4*>(bP[g] + ko);
      *reinterpret_cast<int4*>(&Bs[buf][bOff + g * 32 * BK]) = t;
    }
#endif
  };

  auto wait_async = [&]() {
#ifdef HAVE_ASYNC_LDS
#if __has_builtin(__builtin_amdgcn_s_wait_asynccnt)
    __builtin_amdgcn_s_wait_asynccnt(0);
#else
    asm volatile("s_wait_asynccnt 0" ::: "memory");
#endif
#endif
  };

  // prologue: tile 0
  stage(0, 0);
  wait_async();
  __syncthreads();

  for (int kt = 0; kt < KT2; ++kt) {
    const int cur = kt & 1;
    if ((kt + 1) < KT2) stage(cur ^ 1, (kt + 1) * BK);   // DMA under the WMMAs

    // two K=32 sub-steps per BK=64 slice
#pragma unroll
    for (int kh = 0; kh < 2; ++kh) {
      const int kb = kh * 32;
      // A frags: lanes 0-15 K{0..7,16..23}; lanes 16-31 K{8..15,24..31}
      v16bf afrag[4];
#pragma unroll
      for (int mi = 0; mi < 4; ++mi) {
        int r = wm * 64 + mi * 16 + l16;
        v8bf lo = *reinterpret_cast<const v8bf*>(&As[cur][r * BK + kb + akc]);
        v8bf hv = *reinterpret_cast<const v8bf*>(&As[cur][r * BK + kb + akc + 16]);
        afrag[mi] = __builtin_shufflevector(lo, hv,
            0, 1, 2, 3, 4, 5, 6, 7, 8, 9, 10, 11, 12, 13, 14, 15);
      }
#pragma unroll
      for (int ni = 0; ni < 4; ++ni) {
        int col = wn * 64 + ni * 16 + l16;
        v8bf blo = *reinterpret_cast<const v8bf*>(&Bs[cur][col * BK + kb + bkc]);
        v8bf bhi = *reinterpret_cast<const v8bf*>(&Bs[cur][col * BK + kb + bkc + 16]);
        v16bf bfrag = __builtin_shufflevector(blo, bhi,
            0, 1, 2, 3, 4, 5, 6, 7, 8, 9, 10, 11, 12, 13, 14, 15);
#pragma unroll
        for (int mi = 0; mi < 4; ++mi) {
          acc[mi][ni] = __builtin_amdgcn_wmma_f32_16x16x32_bf16(
              false, afrag[mi], false, bfrag, (short)0, acc[mi][ni], false, false);
        }
      }
    }

    wait_async();        // next tile resident in LDS before anyone reads it
    __syncthreads();
  }

  // Epilogue: bias + relu, gate-weight scale, scatter-add (2 writers/elem,
  // fp add commutative -> bitwise deterministic).
  float bias[4];
#pragma unroll
  for (int ni = 0; ni < 4; ++ni)
    bias[ni] = bexp[e * DDIM + n0 + wn * 64 + ni * 16 + l16];

#pragma unroll
  for (int mi = 0; mi < 4; ++mi) {
#pragma unroll
    for (int v = 0; v < 8; ++v) {
      int r = wm * 64 + mi * 16 + v + (hi ? 8 : 0);  // C layout: VGPR v -> M=v(+8)
      float w = wts[r];
      if (w != 0.f) {
        int tok = toks[r];
#pragma unroll
        for (int ni = 0; ni < 4; ++ni) {
          int col = n0 + wn * 64 + ni * 16 + l16;
          float val = fmaxf(acc[mi][ni][v] + bias[ni], 0.f) * w;
          atomicAdd(out + (size_t)tok * DDIM + col, val);
        }
      }
    }
  }
}

extern "C" void kernel_launch(void* const* d_in, const int* in_sizes, int n_in,
                              void* d_out, int out_size, void* d_ws, size_t ws_size,
                              hipStream_t stream) {
  const float* x  = (const float*)d_in[0];   // [N, D]
  const float* We = (const float*)d_in[1];   // [E, D, D]
  const float* be = (const float*)d_in[2];   // [E, D]
  const float* Wg = (const float*)d_in[3];   // [E, D]
  const float* bg = (const float*)d_in[4];   // [E]
  float* out = (float*)d_out;

  // workspace (~97 MiB): x_bf16 | W_bf16 | counts | tlist | wlist
  char* ws = (char*)d_ws;
  unsigned short* xbf = (unsigned short*)ws;
  size_t off = (size_t)NTOK * DDIM * 2;
  unsigned short* wbf = (unsigned short*)(ws + off);
  off += (size_t)NEXP * DDIM * DDIM * 2;
  int* counts = (int*)(ws + off);   off += 256;
  int* tlist  = (int*)(ws + off);   off += (size_t)NEXP * NTOK * 4;
  float* wlist = (float*)(ws + off);

  (void)hipMemsetAsync(counts, 0, 256, stream);
  (void)hipMemsetAsync(out, 0, (size_t)NTOK * DDIM * sizeof(float), stream);

  {
    int n4 = NTOK * DDIM / 4;
    cvt_f32_bf16<<<(n4 + 255) / 256, 256, 0, stream>>>(x, xbf, n4);
  }
  {
    int n4 = NEXP * DDIM * DDIM / 4;
    cvt_f32_bf16<<<(n4 + 255) / 256, 256, 0, stream>>>(We, wbf, n4);
  }

  moe_gate<<<NTOK / 8, 256, 0, stream>>>(x, Wg, bg, counts, tlist, wlist);

  dim3 grid(DDIM / BN, NTOK / BM, NEXP);
  moe_expert_gemm<<<grid, 256, 0, stream>>>(xbf, wbf, be, counts, tlist, wlist, out);
}
